// Model_27891517620965
// MI455X (gfx1250) — matrix operand order repaired and checked by reading, not compiled, not run on hardware
//
#include <hip/hip_runtime.h>

typedef __bf16 bf16;
typedef __attribute__((ext_vector_type(16))) __bf16 v16bf;
typedef __attribute__((ext_vector_type(8)))  __bf16 v8bf;
typedef __attribute__((ext_vector_type(8)))  float  v8f;
typedef unsigned int u32x4 __attribute__((ext_vector_type(4)));
typedef int          i32x8 __attribute__((ext_vector_type(8)));
typedef int          i32x4 __attribute__((ext_vector_type(4)));

#define B_    64
#define T_    1024
#define IDIM_ 120
#define ID_P  128
#define HDIM_ 320
#define H4_   1280
#define XC_   640
#define K_    72
#define KP_   128
#define LMAX_ 96
#define S_    (2*LMAX_+1)
#define BT_   (B_*T_)
#define NEGF  (-1e30f)

// ---------------------------------------------------------------- async / TDM
// GLOBAL_LOAD_ASYNC_TO_LDS_B128: LDS[lds] = MEM[gaddr], tracked by ASYNCcnt.
// The LDS pointer is passed INTO the asm so the shared object escapes and the
// compiler cannot fold subsequent LDS reads to undef (flat LDS addr[31:0] is
// the LDS byte offset, ISA 10.2).
__device__ __forceinline__ void async_load_b128(void* lds, const void* gaddr) {
  unsigned off = (unsigned)(size_t)lds;
  asm volatile("global_load_async_to_lds_b128 %0, %1, off"
               :: "v"(off), "v"(gaddr), "v"(lds) : "memory");
}
template<int N>
__device__ __forceinline__ void wait_asynccnt() {
  asm volatile("s_wait_asynccnt %0" :: "n"(N) : "memory");
}
template<int N>
__device__ __forceinline__ void wait_tensorcnt() {
  asm volatile("s_wait_tensorcnt %0" :: "n"(N) : "memory");
}

// Tensor Data Mover: 2-D tile (tile_d0 contiguous elems x tile_d1 rows) of bf16
// from global (row stride = stride_elems) into LDS at ldsdst, with optional LDS
// row padding (pad codes per ISA: interval 3 = 16 DWORDs, amount 3 = 4 DWORDs).
__device__ __forceinline__ void tdm_load_2d_bf16(void* ldsdst, const void* gptr,
                                                 unsigned tile_d0, unsigned tile_d1,
                                                 unsigned long long stride_elems,
                                                 unsigned pad_interval, unsigned pad_amount,
                                                 int pad_en)
{
  unsigned long long ga = (unsigned long long)(size_t)gptr;
  unsigned lds_off = (unsigned)(size_t)ldsdst;
  u32x4 g0;
  g0[0] = 1u;                                              // count=1 (valid user D#)
  g0[1] = lds_off;                                         // lds_addr (bytes)
  g0[2] = (unsigned)(ga & 0xffffffffu);                    // global_addr[31:0]
  g0[3] = (unsigned)((ga >> 32) & 0x01ffffffu) | (2u << 30); // addr[56:32] | type=2
  i32x8 g1;
  g1[0] = (int)((1u << 16) |                               // data_size=1 (2 bytes)
                (pad_en ? (1u << 20) : 0u) |
                ((pad_interval & 7u) << 22) |
                ((pad_amount & 127u) << 25));              // workgroup_mask=0
  g1[1] = (int)((tile_d0 & 0xffffu) << 16);                // tensor_dim0[15:0]
  g1[2] = (int)(((tile_d0 >> 16) & 0xffffu) | ((tile_d1 & 0xffffu) << 16)); // d0 hi | d1 lo
  g1[3] = (int)(((tile_d1 >> 16) & 0xffffu) | ((tile_d0 & 0xffffu) << 16)); // d1 hi | tile_dim0
  g1[4] = (int)(tile_d1 & 0xffffu);                        // tile_dim1 | tile_dim2=0
  g1[5] = (int)(unsigned)(stride_elems & 0xffffffffu);     // tensor_dim0_stride lo
  g1[6] = (int)(unsigned)((stride_elems >> 32) & 0xffffu); // stride hi | dim1_stride lo=0
  g1[7] = 0;
  i32x4 g2; g2[0] = 1; g2[1] = 0; g2[2] = 0; g2[3] = 0;    // tensor_dim2=1
  i32x4 g3; g3[0] = 0; g3[1] = 0; g3[2] = 0; g3[3] = 0;
  asm volatile("tensor_load_to_lds %0, %1, %2, %3"
               :: "s"(g0), "s"(g1), "s"(g2), "s"(g3), "v"(ldsdst) : "memory");
}

// ---------------------------------------------------------------- fragments
__device__ __forceinline__ v16bf frag_cat(const bf16* p0, const bf16* p1) {
  v8bf lo = *(const v8bf*)p0;
  v8bf hi = *(const v8bf*)p1;
  return __builtin_shufflevector(lo, hi, 0,1,2,3,4,5,6,7,8,9,10,11,12,13,14,15);
}

// A fragment: 16x32 bf16 tile (ISA 7.12.2 layout)
__device__ __forceinline__ v16bf load_a_frag(const bf16* base, int ld, int m0, int k0) {
  int lane = threadIdx.x & 31;
  const bf16* r = base + (m0 + (lane & 15)) * ld + k0 + ((lane >> 4) << 3);
  return frag_cat(r, r + 16);
}

// B fragment: 32x16, B[k][n] = W[n][k] with W row-major [N][ld]
__device__ __forceinline__ v16bf load_b_frag(const bf16* w, int ld, int n0, int k0) {
  int lane = threadIdx.x & 31;
  const bf16* r = w + (n0 + (lane & 15)) * ld + k0 + ((lane >> 4) << 3);
  return frag_cat(r, r + 16);
}

__device__ __forceinline__ v8f wmma_bf16(v16bf a, v16bf b, v8f c) {
  return __builtin_amdgcn_wmma_f32_16x16x32_bf16(false, a, false, b, (short)0, c,
                                                 false, false);
}

__device__ __forceinline__ float sigf(float x) { return 1.f / (1.f + __expf(-x)); }

// ---------------------------------------------------------------- grid barrier
__device__ __forceinline__ void grid_barrier(int* bar) {
  __syncthreads();
  if (threadIdx.x == 0) {
    __threadfence();
    volatile int* genp = &bar[1];
    int gen = *genp;
    if (atomicAdd(&bar[0], 1) == (int)gridDim.x - 1) {
      bar[0] = 0;
      __threadfence();
      atomicAdd(&bar[1], 1);
    } else {
      while (*genp == gen) { __builtin_amdgcn_s_sleep(2); }
    }
  }
  __syncthreads();
}

// ---------------------------------------------------------------- GEMM (WMMA)
// C[M,N] = A[M,K] * W[N,K]^T + bias[n].  M % 128 == 0, N % 64 == 0, K % 32 == 0.
// A tiles: double-buffered per-lane async B128 copies (ASYNCcnt).
// W tiles: double-buffered TDM descriptors issued by wave 0 (TENSORcnt).
__global__ __launch_bounds__(256)
void gemm_bf16_kernel(const bf16* __restrict__ A, int lda,
                      const bf16* __restrict__ W, int ldw,
                      float* __restrict__ C, int ldc,
                      const float* __restrict__ bias, int Kdim)
{
  // [0,5120): As buf0, [5120,10240): As buf1  (128 rows x 40 bf16)
  // [10240,12800): Ws buf0, [12800,15360): Ws buf1 (64 rows x 40 bf16)
  __shared__ __align__(16) bf16 sb[15360];
  const int m_blk = blockIdx.x * 128;
  const int n_blk = blockIdx.y * 64;
  const int tid  = threadIdx.x;
  const int wave = tid >> 5;
  const int lane = tid & 31;
  const int wm = (wave >> 1) * 32;  // 0,32,64,96
  const int wn = (wave & 1) * 32;   // 0,32
  const int ar = tid >> 1, ac = (tid & 1) * 16;

  v8f acc[2][2];
  #pragma unroll
  for (int i = 0; i < 2; ++i)
    #pragma unroll
    for (int j = 0; j < 2; ++j)
      #pragma unroll
      for (int v = 0; v < 8; ++v) acc[i][j][v] = 0.f;

  auto issueA = [&](int buf, int k0) {
    bf16* dst = sb + buf * 5120 + ar * 40 + ac;
    const bf16* g = &A[(size_t)(m_blk + ar) * lda + k0 + ac];
    async_load_b128(dst, g);
    async_load_b128(dst + 8, g + 8);
  };
  auto issueW = [&](int buf, int k0) {
    if (wave == 0)
      tdm_load_2d_bf16(sb + 10240 + buf * 2560,
                       &W[(size_t)n_blk * ldw + k0],
                       /*tile_d0=*/32, /*tile_d1=*/64,
                       (unsigned long long)ldw,
                       /*pad_interval=16DW*/3, /*pad_amount=4DW*/3, /*pad_en=*/1);
  };

  issueA(0, 0);
  issueW(0, 0);
  int buf = 0;
  for (int k0 = 0; k0 < Kdim; k0 += 32) {
    const bool more = (k0 + 32) < Kdim;
    if (more) {
      issueA(buf ^ 1, k0 + 32);
      issueW(buf ^ 1, k0 + 32);
      wait_asynccnt<2>();                 // previous A tile landed
      if (wave == 0) wait_tensorcnt<1>(); // previous W tile landed
    } else {
      wait_asynccnt<0>();
      if (wave == 0) wait_tensorcnt<0>();
    }
    __syncthreads();

    const bf16* Asb = sb + buf * 5120;
    const bf16* Wsb = sb + 10240 + buf * 2560;
    v16bf a0 = load_a_frag(Asb, 40, wm,      0);
    v16bf a1 = load_a_frag(Asb, 40, wm + 16, 0);
    v16bf b0 = load_b_frag(Wsb, 40, wn,      0);
    v16bf b1 = load_b_frag(Wsb, 40, wn + 16, 0);
    acc[0][0] = wmma_bf16(a0, b0, acc[0][0]);
    acc[0][1] = wmma_bf16(a0, b1, acc[0][1]);
    acc[1][0] = wmma_bf16(a1, b0, acc[1][0]);
    acc[1][1] = wmma_bf16(a1, b1, acc[1][1]);
    __syncthreads();                      // done reading buf before it is reused
    buf ^= 1;
  }

  const int nl = lane & 15;
  const int mb = (lane >> 4) * 8;
  #pragma unroll
  for (int tm = 0; tm < 2; ++tm)
    #pragma unroll
    for (int tn = 0; tn < 2; ++tn) {
      int n = n_blk + wn + tn * 16 + nl;
      float bv = bias ? bias[n] : 0.f;
      #pragma unroll
      for (int v = 0; v < 8; ++v) {
        int m = m_blk + wm + tm * 16 + mb + v;
        C[(size_t)m * ldc + n] = acc[tm][tn][v] + bv;
      }
    }
}

// ---------------------------------------------------------------- LSTM recurrence
// Persistent: 20 WGs x 128 threads; WG owns 16 hidden units (4 gate tiles of 16).
__global__ __launch_bounds__(128)
void lstm_seq_kernel(const float* __restrict__ gates, const bf16* __restrict__ Whh,
                     const int* __restrict__ lengths, bf16* __restrict__ hout,
                     int coloff, int reverse, bf16* __restrict__ hping,
                     int* __restrict__ bar)
{
  __shared__ __align__(16) bf16 hs[64][328];
  const int tid  = threadIdx.x;
  const int wave = tid >> 5;
  const int lane = tid & 31;
  const int n0   = blockIdx.x << 4;  // hidden-unit tile
  const int m0   = wave << 4;        // batch tile (4 waves x 16 = 64)
  const int nl   = lane & 15;
  const int mb   = (lane >> 4) << 3;
  const int n    = n0 + nl;

  v8f c;                               // cell state lives in C-fragment registers
  #pragma unroll
  for (int v = 0; v < 8; ++v) c[v] = 0.f;

  for (int t = 0; t < T_; ++t) {
    const bf16* hr = hping + (t & 1) * (B_ * HDIM_);
    bf16*       hw = hping + ((t + 1) & 1) * (B_ * HDIM_);
    // stage h_prev [64][320] into LDS via async memory->LDS path
    for (int i = tid; i < B_ * 40; i += 128) {
      int r = i / 40, cc = (i - r * 40) << 3;
      async_load_b128(&hs[r][cc], &hr[r * HDIM_ + cc]);
    }
    wait_asynccnt<0>();
    __syncthreads();

    v8f acc[4];
    #pragma unroll
    for (int g = 0; g < 4; ++g)
      #pragma unroll
      for (int v = 0; v < 8; ++v) acc[g][v] = 0.f;

    #pragma unroll 2
    for (int k0 = 0; k0 < HDIM_; k0 += 32) {
      v16bf a = load_a_frag(&hs[0][0], 328, m0, k0);
      #pragma unroll
      for (int g = 0; g < 4; ++g) {
        v16bf b = load_b_frag(Whh, HDIM_, g * HDIM_ + n0, k0);  // L2-resident
        acc[g] = wmma_bf16(a, b, acc[g]);
      }
    }

    #pragma unroll
    for (int v = 0; v < 8; ++v) {
      int mrow = m0 + mb + v;
      const float* grow = gates + ((size_t)mrow * T_ + t) * H4_;
      float xi = grow[0 * HDIM_ + n] + acc[0][v];
      float xf = grow[1 * HDIM_ + n] + acc[1][v];
      float xg = grow[2 * HDIM_ + n] + acc[2][v];
      float xo = grow[3 * HDIM_ + n] + acc[3][v];
      float cv = sigf(xf) * c[v] + sigf(xi) * tanhf(xg);
      c[v] = cv;
      float hv = sigf(xo) * tanhf(cv);
      bf16 hb = (bf16)hv;
      hw[mrow * HDIM_ + n] = hb;
      int len = lengths[mrow];
      int dt = reverse ? (t < len ? len - 1 - t : t) : t;
      hout[((size_t)mrow * T_ + dt) * XC_ + coloff + n] = hb;
      if (t + 1 < T_) __builtin_prefetch(grow + H4_, 0, 1);   // next step's gates
    }
    __threadfence();
    grid_barrier(bar);
  }
}

// ---------------------------------------------------------------- helpers
__global__ void convert_pad_kernel(bf16* __restrict__ dst, const float* __restrict__ src,
                                   int dstRows, int srcRows, int srcK, int dstK)
{
  size_t i = (size_t)blockIdx.x * blockDim.x + threadIdx.x;
  size_t total = (size_t)dstRows * dstK;
  if (i >= total) return;
  int r = (int)(i / dstK), k = (int)(i - (size_t)r * dstK);
  float v = (r < srcRows && k < srcK) ? src[(size_t)r * srcK + k] : 0.f;
  dst[i] = (bf16)v;
}

__global__ void pad_bias_kernel(float* dst, const float* src, int nsrc, int ndst)
{
  int i = blockIdx.x * blockDim.x + threadIdx.x;
  if (i < ndst) dst[i] = (i < nsrc) ? src[i] : 0.f;
}

// dst[b][t][:] = src[b][ t<len ? len-1-t : t ][:], C8 = channels/8
__global__ void reverse_seq_kernel(bf16* __restrict__ dst, const bf16* __restrict__ src,
                                   const int* __restrict__ lengths, int C8)
{
  int row = blockIdx.x;             // b*T + t
  int b = row >> 10, t = row & (T_ - 1);
  int len = lengths[b];
  int st = t < len ? len - 1 - t : t;
  const v8bf* s = (const v8bf*)(src + (((size_t)b * T_ + st) * (size_t)C8) * 8);
  v8bf*       d = (v8bf*)(dst + ((size_t)row * C8) * 8);
  for (int i = threadIdx.x; i < C8; i += blockDim.x) d[i] = s[i];
}

// log_softmax over first 72 of the padded-128 projection row; accumulate denominator
__global__ void logsm_den_kernel(const float* __restrict__ proj, float* __restrict__ logp,
                                 float* __restrict__ den, const int* __restrict__ lengths)
{
  int r = blockIdx.x * blockDim.x + threadIdx.x;
  if (r >= BT_) return;
  int b = r >> 10, t = r & (T_ - 1);
  const float* z = proj + (size_t)r * KP_;
  float m = -3.0e38f;
  for (int k = 0; k < K_; ++k) m = fmaxf(m, z[k]);
  float s = 0.f;
  for (int k = 0; k < K_; ++k) s += __expf(z[k] - m);
  float lse = m + __logf(s);
  float m2 = -3.0e38f;
  float* lp = logp + (size_t)r * K_;
  for (int k = 0; k < K_; ++k) { float v = z[k] - lse; lp[k] = v; m2 = fmaxf(m2, v); }
  float s2 = 0.f;
  for (int k = 0; k < K_; ++k) s2 += __expf(lp[k] - m2);
  float lse2 = m2 + __logf(s2);
  if (t < lengths[b]) atomicAdd(&den[b], lse2);
}

// CTC alpha recursion, one block per batch element, one thread per state
__global__ __launch_bounds__(256)
void ctc_kernel(const float* __restrict__ logp, const int* __restrict__ labels,
                const int* __restrict__ in_lens, const int* __restrict__ lab_lens,
                float* __restrict__ num)
{
  __shared__ int ext[S_];
  __shared__ unsigned char skip[S_];
  __shared__ float a0[S_], a1[S_];
  const int b = blockIdx.x, s = threadIdx.x;
  const int L = lab_lens[b], inlen = in_lens[b];
  const int Smax = 2 * L + 1;
  if (s < S_) ext[s] = (s & 1) ? labels[b * LMAX_ + (s >> 1)] : 0;
  __syncthreads();
  if (s < S_) {
    skip[s] = (unsigned char)((s & 1) && (s >= 2) && (ext[s] != ext[s - 2]));
    float a = NEGF;
    if (s <= 1) a = logp[((size_t)b * T_) * K_ + ext[s]];
    a0[s] = (s < Smax) ? a : NEGF;
  }
  __syncthreads();
  float* cur = a0; float* nxt = a1;
  for (int t = 1; t < T_; ++t) {
    if (s < S_) {
      float an;
      if (t < inlen) {
        float x = cur[s];
        float y = (s >= 1) ? cur[s - 1] : NEGF;
        float z = (s >= 2 && skip[s]) ? cur[s - 2] : NEGF;
        float m = fmaxf(fmaxf(x, y), z);
        float l = m + __logf(__expf(x - m) + __expf(y - m) + __expf(z - m));
        float em = logp[((size_t)b * T_ + t) * K_ + ext[s]];
        an = (s < Smax) ? (l + em) : NEGF;
      } else {
        an = cur[s];
      }
      nxt[s] = an;
    }
    __syncthreads();
    float* tmp = cur; cur = nxt; nxt = tmp;
  }
  if (s == 0) {
    float aL = cur[2 * L], aM = cur[2 * L - 1];
    float m = fmaxf(aL, aM);
    num[b] = m + __logf(__expf(aL - m) + __expf(aM - m));
  }
}

__global__ void final_kernel(const float* den, const float* num, float* out)
{
  __shared__ float red[B_];
  int i = threadIdx.x;
  red[i] = den[i] - 1.1f * num[i];   // 1 + LAMB
  __syncthreads();
  for (int o = 32; o > 0; o >>= 1) {
    if (i < o) red[i] += red[i + o];
    __syncthreads();
  }
  if (i == 0) out[0] = red[0] / (float)B_;
}

// ---------------------------------------------------------------- host
extern "C" void kernel_launch(void* const* d_in, const int* in_sizes, int n_in,
                              void* d_out, int out_size, void* d_ws, size_t ws_size,
                              hipStream_t stream)
{
  (void)in_sizes; (void)n_in; (void)out_size; (void)ws_size;
  const float* logits   = (const float*)d_in[0];
  const int*   labels   = (const int*)  d_in[1];
  const int*   in_lens  = (const int*)  d_in[2];
  const int*   lab_lens = (const int*)  d_in[3];
  const float* Wih0     = (const float*)d_in[4];
  const float* Whh0     = (const float*)d_in[5];
  const float* b0       = (const float*)d_in[6];
  const float* Wih      = (const float*)d_in[7];
  const float* Whh      = (const float*)d_in[8];
  const float* bB       = (const float*)d_in[9];
  const float* W_lin    = (const float*)d_in[10];
  const float* b_lin    = (const float*)d_in[11];
  float* out = (float*)d_out;

  char* ws = (char*)d_ws;
  size_t off = 0;
  auto take = [&](size_t bytes) -> char* {
    char* p = ws + off;
    off = (off + bytes + 255) & ~(size_t)255;
    return p;
  };

  bf16*  x0f    = (bf16*) take((size_t)BT_ * ID_P * 2);
  bf16*  x0r    = (bf16*) take((size_t)BT_ * ID_P * 2);
  bf16*  xcatA  = (bf16*) take((size_t)BT_ * XC_ * 2);
  bf16*  xcatAr = (bf16*) take((size_t)BT_ * XC_ * 2);
  bf16*  xcatB  = (bf16*) take((size_t)BT_ * XC_ * 2);
  bf16*  xcatBr = (bf16*) take((size_t)BT_ * XC_ * 2);
  float* gates  = (float*)take((size_t)BT_ * H4_ * 4);
  bf16*  Wih0b  = (bf16*) take((size_t)2 * H4_ * ID_P * 2);
  bf16*  Whh0b  = (bf16*) take((size_t)2 * H4_ * HDIM_ * 2);
  bf16*  Wihb   = (bf16*) take((size_t)4 * H4_ * XC_ * 2);
  bf16*  Whhb   = (bf16*) take((size_t)4 * H4_ * HDIM_ * 2);
  bf16*  Wlinb  = (bf16*) take((size_t)KP_ * XC_ * 2);
  float* blinp  = (float*)take((size_t)KP_ * 4);
  char*  syncrg = take((size_t)2 * B_ * HDIM_ * 2 + 256);
  bf16*  hping  = (bf16*)syncrg;
  int*   bar    = (int*)(syncrg + (size_t)2 * B_ * HDIM_ * 2);
  float* den    = (float*)take(B_ * 4);
  float* num    = (float*)take(B_ * 4);
  float* proj   = gates;                       // overlay: gates dead by then
  float* logp   = gates + (size_t)BT_ * KP_;   // overlay, disjoint from proj

  auto cgrid = [](size_t nelem) { return dim3((unsigned)((nelem + 255) / 256)); };

  // weight conversion to bf16 (with K/N padding)
  convert_pad_kernel<<<cgrid((size_t)2*H4_*ID_P ), 256, 0, stream>>>(Wih0b, Wih0, 2*H4_, 2*H4_, IDIM_, ID_P);
  convert_pad_kernel<<<cgrid((size_t)2*H4_*HDIM_), 256, 0, stream>>>(Whh0b, Whh0, 2*H4_, 2*H4_, HDIM_, HDIM_);
  convert_pad_kernel<<<cgrid((size_t)4*H4_*XC_  ), 256, 0, stream>>>(Wihb,  Wih,  4*H4_, 4*H4_, XC_,  XC_);
  convert_pad_kernel<<<cgrid((size_t)4*H4_*HDIM_), 256, 0, stream>>>(Whhb,  Whh,  4*H4_, 4*H4_, HDIM_, HDIM_);
  convert_pad_kernel<<<cgrid((size_t)KP_*XC_    ), 256, 0, stream>>>(Wlinb, W_lin, KP_, K_, XC_, XC_);
  pad_bias_kernel<<<1, 128, 0, stream>>>(blinp, b_lin, K_, KP_);

  // layer-0 inputs (batch sort dropped: output is a batch mean, permutation-invariant)
  convert_pad_kernel<<<cgrid((size_t)BT_*ID_P), 256, 0, stream>>>(x0f, logits, BT_, BT_, IDIM_, ID_P);
  reverse_seq_kernel<<<BT_, 64, 0, stream>>>(x0r, x0f, in_lens, ID_P / 8);

  for (int l = 0; l < 3; ++l) {
    const bf16 *inF, *inR; int lda; bf16 *outx, *outr;
    if (l == 0)      { inF = x0f;   inR = x0r;    lda = ID_P; outx = xcatA; outr = xcatAr; }
    else if (l == 1) { inF = xcatA; inR = xcatAr; lda = XC_;  outx = xcatB; outr = xcatBr; }
    else             { inF = xcatB; inR = xcatBr; lda = XC_;  outx = xcatA; outr = nullptr; }

    for (int d = 0; d < 2; ++d) {
      const bf16* wih = (l == 0) ? Wih0b + (size_t)d * H4_ * ID_P
                                 : Wihb  + (size_t)((l - 1) * 2 + d) * H4_ * XC_;
      const bf16* whh = (l == 0) ? Whh0b + (size_t)d * H4_ * HDIM_
                                 : Whhb  + (size_t)((l - 1) * 2 + d) * H4_ * HDIM_;
      const float* bias = (l == 0) ? b0 + (size_t)d * H4_
                                   : bB + (size_t)((l - 1) * 2 + d) * H4_;
      dim3 g(BT_ / 128, H4_ / 64);
      gemm_bf16_kernel<<<g, 256, 0, stream>>>(d ? inR : inF, lda, wih, lda,
                                              gates, H4_, bias, lda);
      hipMemsetAsync(syncrg, 0, (size_t)2 * B_ * HDIM_ * 2 + 8, stream);
      lstm_seq_kernel<<<HDIM_ / 16, 128, 0, stream>>>(gates, whh, in_lens, outx,
                                                      d ? HDIM_ : 0, d, hping, bar);
    }
    if (outr) reverse_seq_kernel<<<BT_, 64, 0, stream>>>(outr, outx, in_lens, XC_ / 8);
  }

  // projection + log-softmax + denominator
  {
    dim3 g(BT_ / 128, KP_ / 64);
    gemm_bf16_kernel<<<g, 256, 0, stream>>>(xcatA, XC_, Wlinb, XC_, proj, KP_, blinp, XC_);
  }
  hipMemsetAsync(den, 0, B_ * 4, stream);
  logsm_den_kernel<<<BT_ / 256, 256, 0, stream>>>(proj, logp, den, in_lens);
  ctc_kernel<<<B_, 256, 0, stream>>>(logp, labels, in_lens, lab_lens, num);
  final_kernel<<<1, B_, 0, stream>>>(den, num, out);
}
